// Glm4MoeAttention_32521492365705
// MI455X (gfx1250) — compile-verified, matrix operand order checked by reading, not run on hardware
//
#include <hip/hip_runtime.h>
#include <hip/hip_bf16.h>

#define HIDDEN   4096
#define N_HEADS  32
#define N_KV     8
#define HEAD_DIM 128
#define ROT_DIM  64
#define T_SEQ    2048
#define Q_SIZE   (N_HEADS * HEAD_DIM)           // 4096
#define KV_SIZE  (N_KV * HEAD_DIM)              // 1024
#define QKV_OUT  (Q_SIZE + 2 * KV_SIZE)         // 6144
#define V_OFF    (Q_SIZE + KV_SIZE)             // 5120
#define EPS_F    1e-6f

typedef __attribute__((ext_vector_type(16))) __bf16    v16bf;
typedef __attribute__((ext_vector_type(8)))  float     v8f;
typedef __attribute__((ext_vector_type(4)))  unsigned  v4u;
typedef __attribute__((ext_vector_type(8)))  unsigned  v8u;

static __device__ __forceinline__ __bf16 f2bf(float f) {
  unsigned u = __float_as_uint(f);
  u += 0x7FFFu + ((u >> 16) & 1u);           // round-to-nearest-even
  unsigned short h = (unsigned short)(u >> 16);
  __bf16 r;
  __builtin_memcpy(&r, &h, sizeof(r));
  return r;
}
static __device__ __forceinline__ float bf2f(__bf16 b) {
  unsigned short u;
  __builtin_memcpy(&u, &b, 2);
  return __uint_as_float((unsigned)u << 16);
}
static __device__ __forceinline__ v8f v8f_zero() {
  v8f z = {0.f, 0.f, 0.f, 0.f, 0.f, 0.f, 0.f, 0.f};
  return z;
}

// Fragment load matching CDNA5 16-bit A/B WMMA layout:
// lanes 0-15: K = k0..k0+7 and k0+16..k0+23 with k0=0; lanes 16-31: k0=8.
static __device__ __forceinline__ v16bf frag16(const __bf16* p, int k0) {
  v16bf r;
#pragma unroll
  for (int i = 0; i < 8; ++i) { r[i] = p[k0 + i]; r[8 + i] = p[k0 + 16 + i]; }
  return r;
}

static __device__ __forceinline__ v8f wmma_bf16(v16bf a, v16bf b, v8f c) {
  return __builtin_amdgcn_wmma_f32_16x16x32_bf16(false, a, false, b, (short)0, c,
                                                 false, false);
}

// ---------------------------------------------------------------------------
// Tensor Data Mover: 2D bf16 tile Global -> LDS, hand-packed D# (§8, 08_async_tensor).
// group0: count=1 | lds_addr | global_addr | type=2.  group1: data_size=2B,
// pad ctrl, tensor_dim0/1, tile_dim0/1, tensor_dim0_stride (elements).
// pad_ctrl produces the padded LDS row stride the WMMA fragment loads expect.
// All operands uniform -> SGPRs; issued by one wave; tracked with TENSORcnt.
// ---------------------------------------------------------------------------
static __device__ __forceinline__ void tdm_load_2d_bf16(
    unsigned lds_off, const __bf16* gptr,
    unsigned tile_d0, unsigned tile_d1,
    unsigned tensor_d0, unsigned tensor_d1,
    unsigned stride_d0, unsigned pad_ctrl) {
  unsigned long long ga = (unsigned long long)(size_t)gptr;
  v4u g0;
  g0[0] = 1u;                                             // count=1, user D#
  g0[1] = lds_off;                                        // LDS byte address
  g0[2] = (unsigned)(ga & 0xFFFFFFFFu);                   // global_addr[31:0]
  g0[3] = (unsigned)((ga >> 32) & 0x1FFFFFFu) | (2u << 30); // addr[56:32]|type=2
  v8u g1;
  g1[0] = (1u << 16) | pad_ctrl;                          // data_size=2B | pad
  g1[1] = (tensor_d0 & 0xFFFFu) << 16;                    // tensor_dim0[15:0]
  g1[2] = (tensor_d0 >> 16) | ((tensor_d1 & 0xFFFFu) << 16);
  g1[3] = (tensor_d1 >> 16) | (tile_d0 << 16);            // tile_dim0
  g1[4] = tile_d1;                                        // tile_dim1 (dim2=0)
  g1[5] = stride_d0;                                      // dim0 stride lo32
  g1[6] = 0;
  g1[7] = 0;
  asm volatile("tensor_load_to_lds %0, %1" :: "s"(g0), "s"(g1) : "memory");
}
// pad: row = 32 elems (16 DW, code 3) + 8 elems (4 DW, code 3) -> stride 40
#define PAD_ROW32  ((1u << 20) | (3u << 22) | (3u << 25))
// pad: row = 128 elems (64 DW, code 5) + 8 elems (4 DW, code 3) -> stride 136
#define PAD_ROW128 ((1u << 20) | (5u << 22) | (3u << 25))

// ---------------------------------------------------------------------------
// fp32 -> bf16 bulk convert (one-time; makes weights L2-resident at half size)
// ---------------------------------------------------------------------------
__global__ __launch_bounds__(256) void cvt_f32_bf16(const float* __restrict__ in,
                                                    __bf16* __restrict__ out, int n) {
  int i = (blockIdx.x * 256 + threadIdx.x) * 4;
  if (i < n) {
    float4 v = *(const float4*)(in + i);
    out[i]     = f2bf(v.x);
    out[i + 1] = f2bf(v.y);
    out[i + 2] = f2bf(v.z);
    out[i + 3] = f2bf(v.w);
  }
}

// ---------------------------------------------------------------------------
// GEMM: C[M,N] = A[M,K] * B[K,N]; A,B bf16; C fp32 or bf16.
// Block 256 threads (8 waves), tile 128x128, K-step 32.
// A tile staged by TDM (padded rows, stride 40 elems); B tile staged manually
// N-major (needs transpose). Fragments via ds_load_b128 pairs.
// ---------------------------------------------------------------------------
template <bool BF16OUT>
__global__ __launch_bounds__(256) void gemm_bf16_wmma(
    const __bf16* __restrict__ A, const __bf16* __restrict__ B,
    void* __restrict__ Cv, int M, int N, int K) {
  __shared__ __bf16 As[128 * 40];   // [m][k], stride 40 (TDM-padded)
  __shared__ __bf16 Bs[128 * 40];   // [n][k] (transposed), stride 40

  const int tid  = threadIdx.x;
  const int wave = tid >> 5, lane = tid & 31;
  const int wr = wave & 1, wc = wave >> 1;
  const int lr = lane & 15, half = (lane < 16) ? 0 : 8;
  const int m0 = blockIdx.y * 128, n0 = blockIdx.x * 128;
  const unsigned as_off = (unsigned)(size_t)(&As[0]);

  v8f acc[4][2];
#pragma unroll
  for (int i = 0; i < 4; ++i)
#pragma unroll
    for (int j = 0; j < 2; ++j) acc[i][j] = v8f_zero();

  for (int k0 = 0; k0 < K; k0 += 32) {
    __syncthreads();                          // prior-iter LDS reads done
    if (wave == 0)                            // async DMA: A tile 128x32
      tdm_load_2d_bf16(as_off, A + (size_t)m0 * K + k0,
                       /*tile*/ 32, 128, /*tensor*/ (unsigned)(K - k0),
                       (unsigned)(M - m0), (unsigned)K, PAD_ROW32);
#pragma unroll
    for (int i = 0; i < 16; ++i) {            // B tile 32x128, store N-major
      int kk = (tid >> 7) + i * 2, n = tid & 127;
      Bs[n * 40 + kk] = B[(size_t)(k0 + kk) * N + n0 + n];
    }
    if (wave == 0) __builtin_amdgcn_s_wait_tensorcnt(0);
    __syncthreads();                          // A (TDM) + B (ds) published

    v16bf bf0 = frag16(&Bs[(wc * 32 + lr) * 40], half);
    v16bf bf1 = frag16(&Bs[(wc * 32 + 16 + lr) * 40], half);
#pragma unroll
    for (int i = 0; i < 4; ++i) {
      v16bf af = frag16(&As[(wr * 64 + i * 16 + lr) * 40], half);
      acc[i][0] = wmma_bf16(af, bf0, acc[i][0]);
      acc[i][1] = wmma_bf16(af, bf1, acc[i][1]);
    }
  }

  // C layout: VGPR r -> row (half + r), col = lane & 15
  const size_t cbase =
      (size_t)(m0 + wr * 64 + half) * N + (size_t)(n0 + wc * 32 + lr);
#pragma unroll
  for (int i = 0; i < 4; ++i)
#pragma unroll
    for (int j = 0; j < 2; ++j)
#pragma unroll
      for (int r = 0; r < 8; ++r) {
        int off = (i * 16 + r) * N + j * 16;
        if (BF16OUT) ((__bf16*)Cv)[cbase + off] = f2bf(acc[i][j][r]);
        else         ((float*)Cv)[cbase + off]  = acc[i][j][r];
      }
}

// ---------------------------------------------------------------------------
// Per-(token, head) RMSNorm + RoPE, bf16 in/out, fp32 math.
// grid.x = head (0..31 -> Q, 32..39 -> K), grid.y = token. 128 threads.
// ---------------------------------------------------------------------------
__global__ __launch_bounds__(128) void normrope_kernel(
    const __bf16* __restrict__ qkv, const int* __restrict__ positions,
    const float* __restrict__ qw, const float* __restrict__ kw,
    __bf16* __restrict__ qn, __bf16* __restrict__ kn) {
  __shared__ float red[128];
  __shared__ float xs[128];
  const int d = threadIdx.x;
  const int head = blockIdx.x;
  const int t = blockIdx.y;
  const bool isq = head < N_HEADS;
  const int off = isq ? head * HEAD_DIM : Q_SIZE + (head - N_HEADS) * HEAD_DIM;

  float x = bf2f(qkv[(size_t)t * QKV_OUT + off + d]);
  red[d] = x * x;
  __syncthreads();
  for (int s = 64; s > 0; s >>= 1) {
    if (d < s) red[d] += red[d + s];
    __syncthreads();
  }
  float rs = rsqrtf(red[0] * (1.0f / HEAD_DIM) + EPS_F);
  float w = isq ? qw[d] : kw[d];
  float xn = x * rs * w;
  xs[d] = xn;
  __syncthreads();

  float out = xn;
  if (d < ROT_DIM) {
    int i = (d < 32) ? d : (d - 32);
    // inv_freq = ROPE_BASE^(-2i/ROT_DIM); ln(1e6) = 13.815510557964274
    float inv_freq = __expf(-(2.0f * (float)i / (float)ROT_DIM) * 13.815510557964274f);
    float fr = (float)positions[t] * inv_freq;
    float c = __cosf(fr), s = __sinf(fr);
    float x1 = xs[i], x2 = xs[i + 32];
    out = (d < 32) ? (x1 * c - x2 * s) : (x2 * c + x1 * s);
  }
  if (isq) qn[(size_t)t * Q_SIZE + head * HEAD_DIM + d] = f2bf(out);
  else     kn[(size_t)t * KV_SIZE + (head - N_HEADS) * HEAD_DIM + d] = f2bf(out);
}

// ---------------------------------------------------------------------------
// Flash-style causal attention, all-bf16 operands, fp32 online softmax.
// Block = 128 threads (4 waves) handles (head, 64 query rows); 32 keys/iter.
// K tile staged by TDM (padded rows -> stride 136); V staged transposed.
// ---------------------------------------------------------------------------
__global__ __launch_bounds__(128) void attn_kernel(
    const __bf16* __restrict__ qn, const __bf16* __restrict__ kn,
    const __bf16* __restrict__ qkv, __bf16* __restrict__ out) {
  __shared__ __bf16 Klds[32 * 136];       // [key][dim], stride 136 (TDM-padded)
  __shared__ __bf16 Vlds[128 * 40];       // [dim][key] (transposed), stride 40
  __shared__ __bf16 Plds[4 * 16 * 40];    // per-wave 16x32 probs, stride 40

  const int tid  = threadIdx.x;
  const int wave = tid >> 5, lane = tid & 31;
  const int lr = lane & 15, half = (lane < 16) ? 0 : 8;
  const int h = blockIdx.x;
  const int q0 = blockIdx.y * 64;
  const int kvh = h / (N_HEADS / N_KV);
  const float scale = 0.08838834764831845f;   // 1/sqrt(128)
  const unsigned klds_off = (unsigned)(size_t)(&Klds[0]);

  const __bf16* qrow = qn + (size_t)(q0 + wave * 16 + lr) * Q_SIZE + h * HEAD_DIM;
  v16bf qf[4];
#pragma unroll
  for (int c = 0; c < 4; ++c) qf[c] = frag16(qrow + c * 32, half);

  v8f o[8];
#pragma unroll
  for (int j = 0; j < 8; ++j) o[j] = v8f_zero();
  float m[8], sum[8];
#pragma unroll
  for (int r = 0; r < 8; ++r) { m[r] = -1e30f; sum[r] = 0.0f; }

  __bf16* Pw = &Plds[wave * 16 * 40];
  const __bf16* vbase = qkv + V_OFF + kvh * HEAD_DIM;
  const int kend = q0 + 64;                 // uniform per block -> barriers legal

  for (int kt = 0; kt < kend; kt += 32) {
    __syncthreads();
    if (wave == 0)                          // async DMA: K tile 32x128
      tdm_load_2d_bf16(klds_off, kn + (size_t)kt * KV_SIZE + kvh * HEAD_DIM,
                       /*tile*/ 128, 32, /*tensor*/ 128u,
                       (unsigned)(T_SEQ - kt) * N_KV, (unsigned)KV_SIZE,
                       PAD_ROW128);
#pragma unroll 4
    for (int i = 0; i < 32; ++i)            // V tile transposed [dim][key]
      Vlds[tid * 40 + i] = vbase[(size_t)(kt + i) * QKV_OUT + tid];
    if (wave == 0) __builtin_amdgcn_s_wait_tensorcnt(0);
    __syncthreads();

    // scores: S[16x32] = Q[16x128] * K^T[128x32]
    v8f c0 = v8f_zero(), c1 = v8f_zero();
#pragma unroll
    for (int c = 0; c < 4; ++c) {
      v16bf b0 = frag16(&Klds[lr * 136 + c * 32], half);        // keys 0..15
      v16bf b1 = frag16(&Klds[(16 + lr) * 136 + c * 32], half); // keys 16..31
      c0 = wmma_bf16(qf[c], b0, c0);
      c1 = wmma_bf16(qf[c], b1, c1);
    }

    // online softmax (row's 16 cols live across 16 lanes of each half-wave)
#pragma unroll
    for (int r = 0; r < 8; ++r) {
      int row  = q0 + wave * 16 + half + r;
      bool k0m = (kt + lr) > row;
      bool k1m = (kt + 16 + lr) > row;
      float s0 = k0m ? -1e30f : c0[r] * scale;
      float s1 = k1m ? -1e30f : c1[r] * scale;
      float tmax = fmaxf(s0, s1);
#pragma unroll
      for (int offd = 1; offd < 16; offd <<= 1)
        tmax = fmaxf(tmax, __shfl_xor(tmax, offd, 16));
      float mn = fmaxf(m[r], tmax);
      float f = __expf(m[r] - mn);
      m[r] = mn;
      sum[r] *= f;
#pragma unroll
      for (int j = 0; j < 8; ++j) o[j][r] *= f;
      float p0 = k0m ? 0.0f : __expf(s0 - mn);
      float p1 = k1m ? 0.0f : __expf(s1 - mn);
      sum[r] += p0 + p1;
      Pw[(half + r) * 40 + lr]      = f2bf(p0);   // C-layout -> A-layout via LDS
      Pw[(half + r) * 40 + 16 + lr] = f2bf(p1);
    }
    __syncthreads();

    // O[16x128] += P[16x32] * V[32x128]
    v16bf pa = frag16(&Pw[lr * 40], half);
#pragma unroll
    for (int j = 0; j < 8; ++j) {
      v16bf vb = frag16(&Vlds[(j * 16 + lr) * 40], half);
      o[j] = wmma_bf16(pa, vb, o[j]);
    }
  }

  // finalize: reduce row sums across 16 lanes, normalize, store bf16
#pragma unroll
  for (int r = 0; r < 8; ++r) {
#pragma unroll
    for (int offd = 1; offd < 16; offd <<= 1)
      sum[r] += __shfl_xor(sum[r], offd, 16);
  }
#pragma unroll
  for (int r = 0; r < 8; ++r) {
    float inv = 1.0f / sum[r];
    int row = q0 + wave * 16 + half + r;
#pragma unroll
    for (int j = 0; j < 8; ++j)
      out[(size_t)row * Q_SIZE + h * HEAD_DIM + j * 16 + lr] =
          f2bf(o[j][r] * inv);
  }
}

// ---------------------------------------------------------------------------
extern "C" void kernel_launch(void* const* d_in, const int* in_sizes, int n_in,
                              void* d_out, int out_size, void* d_ws, size_t ws_size,
                              hipStream_t stream) {
  const float* hidden    = (const float*)d_in[0];
  const int*   positions = (const int*)d_in[1];
  const float* w_qkv     = (const float*)d_in[2];
  const float* w_o       = (const float*)d_in[3];
  const float* q_norm_w  = (const float*)d_in[4];
  const float* k_norm_w  = (const float*)d_in[5];
  float* out = (float*)d_out;

  // workspace: all-bf16 staging (~163 MB)
  __bf16* hb   = (__bf16*)d_ws;
  __bf16* wqkb = hb   + (size_t)T_SEQ * HIDDEN;
  __bf16* wob  = wqkb + (size_t)HIDDEN * QKV_OUT;
  __bf16* qkvb = wob  + (size_t)HIDDEN * HIDDEN;
  __bf16* qnb  = qkvb + (size_t)T_SEQ * QKV_OUT;
  __bf16* knb  = qnb  + (size_t)T_SEQ * Q_SIZE;
  __bf16* aob  = knb  + (size_t)T_SEQ * KV_SIZE;

  int nh = T_SEQ * HIDDEN, nq = HIDDEN * QKV_OUT, no = HIDDEN * HIDDEN;
  cvt_f32_bf16<<<nh / 1024, 256, 0, stream>>>(hidden, hb, nh);
  cvt_f32_bf16<<<nq / 1024, 256, 0, stream>>>(w_qkv, wqkb, nq);
  cvt_f32_bf16<<<no / 1024, 256, 0, stream>>>(w_o, wob, no);

  gemm_bf16_wmma<true><<<dim3(QKV_OUT / 128, T_SEQ / 128), 256, 0, stream>>>(
      hb, wqkb, qkvb, T_SEQ, QKV_OUT, HIDDEN);

  normrope_kernel<<<dim3(N_HEADS + N_KV, T_SEQ), 128, 0, stream>>>(
      qkvb, positions, q_norm_w, k_norm_w, qnb, knb);

  attn_kernel<<<dim3(N_HEADS, T_SEQ / 64), 128, 0, stream>>>(qnb, knb, qkvb, aob);

  gemm_bf16_wmma<false><<<dim3(HIDDEN / 128, T_SEQ / 128), 256, 0, stream>>>(
      aob, wob, out, T_SEQ, HIDDEN, HIDDEN);
}
